// SupConLoss_41566693491158
// MI455X (gfx1250) — matrix operand order repaired and checked by reading, not compiled, not run on hardware
//
#include <hip/hip_runtime.h>
#include <math.h>

typedef __attribute__((ext_vector_type(16))) _Float16 v16h;
typedef __attribute__((ext_vector_type(8)))  float    v8f;

#define BROWS   4096
#define DIM     256
#define NTOT    8192
#define INV_T   14.285714285714286f   // 1 / 0.07, also the fixed LSE shift

#define SPLITS          8
#define COLS_PER_SPLIT  (NTOT / SPLITS)   // 1024
#define COLS_PER_STAGE  64
#define ROWS_PER_BLOCK  64
#define THREADS         128
#define LDS_STRIDE      264   // 256 + 8 halves pad -> conflict-free b128 LDS reads

// ---------------------------------------------------------------------------
// Kernel 1: L2-normalize rows of z_i / z_j into fp16 z_hat (8192 x 256)
// ---------------------------------------------------------------------------
__global__ __launch_bounds__(256) void normalize_rows(
    const float* __restrict__ zi, const float* __restrict__ zj,
    _Float16* __restrict__ zh) {
  const int wave = threadIdx.x >> 5;
  const int lane = threadIdx.x & 31;
  const int row  = blockIdx.x * 8 + wave;
  if (row >= NTOT) return;
  const float* src = (row < BROWS) ? (zi + (size_t)row * DIM)
                                   : (zj + (size_t)(row - BROWS) * DIM);
  float4 a = *(const float4*)(src + lane * 8);
  float4 b = *(const float4*)(src + lane * 8 + 4);
  float ss = a.x * a.x + a.y * a.y + a.z * a.z + a.w * a.w +
             b.x * b.x + b.y * b.y + b.z * b.z + b.w * b.w;
#pragma unroll
  for (int off = 16; off > 0; off >>= 1) ss += __shfl_xor(ss, off, 32);
  const float s = 1.0f / fmaxf(sqrtf(ss), 1e-12f);
  union { _Float16 h[8]; uint4 u; } o;
  o.h[0] = (_Float16)(a.x * s); o.h[1] = (_Float16)(a.y * s);
  o.h[2] = (_Float16)(a.z * s); o.h[3] = (_Float16)(a.w * s);
  o.h[4] = (_Float16)(b.x * s); o.h[5] = (_Float16)(b.y * s);
  o.h[6] = (_Float16)(b.z * s); o.h[7] = (_Float16)(b.w * s);
  *(uint4*)(zh + (size_t)row * DIM + lane * 8) = o.u;
}

// ---------------------------------------------------------------------------
// Kernel 2: partial fused sim + masked accumulators over a column split.
// Since sim/T is bounded by 1/T, logsumexp uses a FIXED shift (1/T):
// partials (pos_sum, sum_exp, pos_cnt) are additive across splits.
// Block = 4 waves (64 rows); grid = (128 row-blocks, 8 column-splits).
// Column tiles are staged into LDS via async global->LDS DMA (ASYNCcnt).
// ---------------------------------------------------------------------------
__global__ __launch_bounds__(THREADS) void supcon_partial(
    const _Float16* __restrict__ zh, const int* __restrict__ labels,
    float4* __restrict__ part) {
  __shared__ __align__(16) _Float16 lds[COLS_PER_STAGE * LDS_STRIDE];

  const int tid   = threadIdx.x;
  const int wave  = tid >> 5;
  const int lane  = tid & 31;
  const int hi    = lane >> 4;   // lane group (C layout: M = r + 8*hi)
  const int l16   = lane & 15;   // C layout: N = l16
  const int m0    = blockIdx.x * ROWS_PER_BLOCK + wave * 16;
  const int split = blockIdx.y;

  int lab_row[8];
#pragma unroll
  for (int r = 0; r < 8; ++r)
    lab_row[r] = labels[(m0 + hi * 8 + r) & (BROWS - 1)];

  // A fragments (16x32 fp16, ISA per-lane layout), resident for whole kernel
  v16h afrag[8];
  {
    const _Float16* arow = zh + (size_t)(m0 + l16) * DIM;
#pragma unroll
    for (int kk = 0; kk < 8; ++kk) {
      union { uint4 u[2]; v16h v; } t;
      t.u[0] = *(const uint4*)(arow + kk * 32 + hi * 8);
      t.u[1] = *(const uint4*)(arow + kk * 32 + hi * 8 + 16);
      afrag[kk] = t.v;
    }
  }

  float pos[8], se[8], cnt[8];
#pragma unroll
  for (int r = 0; r < 8; ++r) { pos[r] = 0.0f; se[r] = 0.0f; cnt[r] = 0.0f; }

  // async-stage addressing: each thread owns half of one of the 64 rows
  const unsigned stage_row = (unsigned)(tid >> 1);
  const unsigned stage_p   = (unsigned)(tid & 1);
  const unsigned lbase =
      (unsigned)(size_t)(lds + stage_row * LDS_STRIDE + stage_p * 128);

  const int jb0 = split * COLS_PER_SPLIT;
#pragma unroll 1
  for (int jb = jb0; jb < jb0 + COLS_PER_SPLIT; jb += COLS_PER_STAGE) {
    // ---- async DMA: 64 zh rows (32 KB) -> LDS, 16 B per issue per lane ----
    {
      const unsigned gvoff =
          (unsigned)(((unsigned)(jb) + stage_row) * DIM + stage_p * 128) *
          (unsigned)sizeof(_Float16);
#pragma unroll
      for (int q = 0; q < 16; ++q) {
        asm volatile("global_load_async_to_lds_b128 %0, %1, %2"
                     :: "v"(lbase + q * 16), "v"(gvoff + q * 16), "s"(zh)
                     : "memory");
      }
    }
    // column labels for the 4 sub-tiles of this stage (register resident)
    int lab_c0 = labels[(jb +  0 + l16) & (BROWS - 1)];
    int lab_c1 = labels[(jb + 16 + l16) & (BROWS - 1)];
    int lab_c2 = labels[(jb + 32 + l16) & (BROWS - 1)];
    int lab_c3 = labels[(jb + 48 + l16) & (BROWS - 1)];

    asm volatile("s_wait_asynccnt 0x0" ::: "memory");
    __syncthreads();

#pragma unroll
    for (int jt = 0; jt < COLS_PER_STAGE; jt += 16) {
      // B fragments first (all 16 ds_load_b128), then the chained WMMAs
      const _Float16* brow = lds + (jt + l16) * LDS_STRIDE + hi * 8;
      v16h bf[8];
#pragma unroll
      for (int kk = 0; kk < 8; ++kk) {
        union { uint4 u[2]; v16h v; } t;
        t.u[0] = *(const uint4*)(brow + kk * 32);
        t.u[1] = *(const uint4*)(brow + kk * 32 + 16);
        bf[kk] = t.v;
      }
      v8f c = {};
#pragma unroll
      for (int kk = 0; kk < 8; ++kk)
        c = __builtin_amdgcn_wmma_f32_16x16x32_f16(
            false, afrag[kk], false, bf[kk], (short)0, c, false, false);

      const int jg = jb + jt + l16;
      const int lc = (jt == 0) ? lab_c0 : (jt == 16) ? lab_c1
                   : (jt == 32) ? lab_c2 : lab_c3;
      // branchless masked accumulation (no exec divergence)
#pragma unroll
      for (int r = 0; r < 8; ++r) {
        const float sv   = c[r] * INV_T;
        const bool  same = (lc == lab_row[r]);
        const bool  pv   = same && (jg != (m0 + hi * 8 + r));
        const float e    = __expf(sv - INV_T);   // in (0, 1]: safe fixed shift
        se[r]  += same ? 0.0f : e;
        pos[r] += pv ? sv : 0.0f;
        cnt[r] += pv ? 1.0f : 0.0f;
      }
    }
    __syncthreads();
  }

  // merge the 16 N-partials per row across each 16-lane group; store partials
#pragma unroll
  for (int r = 0; r < 8; ++r) {
    float p = pos[r], e = se[r], cn = cnt[r];
#pragma unroll
    for (int off = 1; off < 16; off <<= 1) {
      p  += __shfl_xor(p, off, 32);
      e  += __shfl_xor(e, off, 32);
      cn += __shfl_xor(cn, off, 32);
    }
    if (l16 == 0)
      part[(size_t)(m0 + hi * 8 + r) * SPLITS + split] =
          make_float4(p, e, cn, 0.0f);
  }
}

// ---------------------------------------------------------------------------
// Kernel 3: combine split partials into per-row loss
// ---------------------------------------------------------------------------
__global__ __launch_bounds__(256) void combine_rows(
    const float4* __restrict__ part, float* __restrict__ row_loss) {
  const int row = blockIdx.x * 256 + threadIdx.x;
  if (row >= NTOT) return;
  float p = 0.0f, e = 0.0f, cn = 0.0f;
#pragma unroll
  for (int s = 0; s < SPLITS; ++s) {
    const float4 v = part[(size_t)row * SPLITS + s];
    p += v.x; e += v.y; cn += v.z;
  }
  float li = 0.0f;
  if (cn > 0.0f && e > 0.0f) {
    const float lse = __logf(e) + INV_T;   // undo fixed shift
    const float d   = lse - p;
    li = (d > 30.0f) ? d : log1pf(__expf(d));
  }
  row_loss[row] = li;
}

// ---------------------------------------------------------------------------
// Kernel 4: deterministic scalar reduction  loss = sum(row_loss) / N
// ---------------------------------------------------------------------------
__global__ __launch_bounds__(256) void final_reduce(
    const float* __restrict__ row_loss, float* __restrict__ out) {
  __shared__ float sbuf[256];
  float s = 0.0f;
  for (int i = threadIdx.x; i < NTOT; i += 256) s += row_loss[i];
  sbuf[threadIdx.x] = s;
  __syncthreads();
  for (int off = 128; off > 0; off >>= 1) {
    if (threadIdx.x < off) sbuf[threadIdx.x] += sbuf[threadIdx.x + off];
    __syncthreads();
  }
  if (threadIdx.x == 0) out[0] = sbuf[0] / (float)NTOT;
}

extern "C" void kernel_launch(void* const* d_in, const int* in_sizes, int n_in,
                              void* d_out, int out_size, void* d_ws, size_t ws_size,
                              hipStream_t stream) {
  const float* zi     = (const float*)d_in[0];
  const float* zj     = (const float*)d_in[1];
  const int*   labels = (const int*)d_in[2];

  char* ws = (char*)d_ws;
  _Float16* zh       = (_Float16*)ws;                      // 4 MB
  float4*   part     = (float4*)(ws + (size_t)NTOT * DIM * sizeof(_Float16)); // 1 MB
  float*    row_loss = (float*)(ws + (size_t)NTOT * DIM * sizeof(_Float16)
                                   + (size_t)NTOT * SPLITS * sizeof(float4));
  float*    out      = (float*)d_out;

  normalize_rows<<<NTOT / 8, 256, 0, stream>>>(zi, zj, zh);
  supcon_partial<<<dim3(NTOT / ROWS_PER_BLOCK, SPLITS), THREADS, 0, stream>>>(
      zh, labels, part);
  combine_rows<<<NTOT / 256, 256, 0, stream>>>(part, row_loss);
  final_reduce<<<1, 256, 0, stream>>>(row_loss, out);
}
